// FlashAttentionMini_69827578298932
// MI455X (gfx1250) — compile-verified, matrix operand order checked by reading, not run on hardware
//
#include <hip/hip_runtime.h>
#include <hip/hip_bf16.h>

typedef __attribute__((ext_vector_type(16))) _Float16 v16h;
typedef __attribute__((ext_vector_type(8)))  _Float16 v8h;
typedef __attribute__((ext_vector_type(4)))  _Float16 v4h;
typedef __attribute__((ext_vector_type(8)))  float    v8f;

#define BB 2
#define SS 2048
#define EE 1024
#define HH 16
#define HD 64
#define MM (BB*SS)   // 4096 rows

// ---------- fragment helpers ----------
static __device__ __forceinline__ v16h make_frag(v8h lo, v8h hi) {
  return __builtin_shufflevector(lo, hi, 0,1,2,3,4,5,6,7,8,9,10,11,12,13,14,15);
}

// Load a 16x32 f16 operand tile from a row-major source with leading dim `ld`
// (halves). Per ISA 7.12.2: lane L (<16) holds row M=L, K={0..7, 16..23};
// lane L (>=16) holds row M=L-16, K={8..15, 24..31}. The B operand uses the
// mirrored layout (lane holds *column* N, same K pattern), so the same loader
// serves B when `p` points at the matrix whose rows are the output columns.
static __device__ __forceinline__ v16h load_frag(const _Float16* p, int ld) {
  const int lane = threadIdx.x & 31;
  const int r  = lane & 15;
  const int k0 = (lane >> 4) << 3;     // 0 or 8
  const _Float16* q = p + (size_t)r * (size_t)ld + k0;
  v8h lo = *(const v8h*)(q);
  v8h hi = *(const v8h*)(q + 16);
  return make_frag(lo, hi);
}

static __device__ __forceinline__ v8f wmma16(v16h a, v16h b, v8f c) {
  return __builtin_amdgcn_wmma_f32_16x16x32_f16(false, a, false, b, (short)0, c, false, false);
}

static __device__ __forceinline__ float rmax16(float x) {
  x = fmaxf(x, __shfl_xor(x, 1, 32));
  x = fmaxf(x, __shfl_xor(x, 2, 32));
  x = fmaxf(x, __shfl_xor(x, 4, 32));
  x = fmaxf(x, __shfl_xor(x, 8, 32));
  return x;
}
static __device__ __forceinline__ float rsum16(float x) {
  x += __shfl_xor(x, 1, 32);
  x += __shfl_xor(x, 2, 32);
  x += __shfl_xor(x, 4, 32);
  x += __shfl_xor(x, 8, 32);
  return x;
}

// gfx1250 async copy global->LDS, tracked by ASYNCcnt (CDNA5 ISA).
// Same INST_OFFSET is added to the LDS and global addresses.
#define ASYNC_LD_B128(dst, src, offstr)                                        \
  asm volatile("global_load_async_to_lds_b128 %0, %1, off offset:" offstr      \
               : : "v"(dst), "v"(src) : "memory")
#define WAIT_ASYNC0() asm volatile("s_wait_asynccnt 0x0" ::: "memory")

// ---------- f32 -> f16 convert ----------
__global__ void cvt_f32_f16_kernel(const float* __restrict__ src,
                                   _Float16* __restrict__ dst, int n4) {
  int i = blockIdx.x * blockDim.x + threadIdx.x;
  if (i < n4) {
    float4 f = ((const float4*)src)[i];
    v4h h;
    h[0] = (_Float16)f.x; h[1] = (_Float16)f.y;
    h[2] = (_Float16)f.z; h[3] = (_Float16)f.w;
    ((v4h*)dst)[i] = h;
  }
}

// ---------- generic GEMM: C = A(f16)[MM x EE] @ W(f16)[EE x EE]^T + bias ----------
// One wave computes a 32(M) x 64(N) tile: 8 WMMAs per K-step, A frags reused 4x,
// B frags reused 2x. MODE 0: f16 [B,H,S,HD] (scaled); 1: f16 [B,H,HD,S]; 2: f32 row-major.
template <int MODE>
__global__ __launch_bounds__(32) void gemm_kernel(
    const _Float16* __restrict__ A, const _Float16* __restrict__ W,
    const float* __restrict__ bias, _Float16* __restrict__ outh,
    float* __restrict__ outf, float scale)
{
  const int m0   = blockIdx.x * 32;
  const int n0   = blockIdx.y * 64;
  const int lane = threadIdx.x & 31;
  const int r    = lane & 15;
  const int hf   = lane >> 4;

  v8f acc[2][4];
#pragma unroll
  for (int mt = 0; mt < 2; ++mt)
#pragma unroll
    for (int nd = 0; nd < 4; ++nd)
#pragma unroll
      for (int v = 0; v < 8; ++v) acc[mt][nd][v] = 0.0f;

  for (int k = 0; k < EE; k += 32) {
    if (k + 64 < EE) {  // prefetch one cacheline ahead on the streamed rows
      __builtin_prefetch(A + (size_t)(m0 + r) * EE + k + 64, 0, 3);
      __builtin_prefetch(W + (size_t)(n0 + r) * EE + k + 64, 0, 3);
    }
    v16h a0 = load_frag(A + (size_t)m0 * EE + k, EE);
    v16h a1 = load_frag(A + (size_t)(m0 + 16) * EE + k, EE);
#pragma unroll
    for (int nd = 0; nd < 4; ++nd) {
      v16h b = load_frag(W + (size_t)(n0 + nd * 16) * EE + k, EE);
      acc[0][nd] = wmma16(a0, b, acc[0][nd]);
      acc[1][nd] = wmma16(a1, b, acc[1][nd]);
    }
  }

#pragma unroll
  for (int mt = 0; mt < 2; ++mt) {
#pragma unroll
    for (int nd = 0; nd < 4; ++nd) {
      const int n = n0 + nd * 16 + r;
      const float bv = bias[n];
#pragma unroll
      for (int v = 0; v < 8; ++v) {
        const int m = m0 + mt * 16 + v + 8 * hf;
        const float val = (acc[mt][nd][v] + bv) * scale;
        if (MODE == 2) {
          outf[(size_t)m * EE + n] = val;
        } else {
          const int b_ = m >> 11;
          const int s  = m & (SS - 1);
          const int h  = n >> 6;
          const int d  = n & (HD - 1);
          if (MODE == 0)
            outh[(((size_t)(b_ * HH + h)) * SS + s) * HD + d] = (_Float16)val;
          else
            outh[(((size_t)(b_ * HH + h)) * HD + d) * SS + s] = (_Float16)val;
        }
      }
    }
  }
}

// ---------- flash attention ----------
// One wave per (b, h, 32-row query block). BLK==tile -> mask is the loop bound.
// K/V chunks are double-buffered in LDS via GLOBAL_LOAD_ASYNC_TO_LDS_B128:
// chunk j+1 streams in while chunk j feeds WMMA + online softmax.
// LDS rows padded (K:144B, V/P:80B) so ds_load_b128 frag reads avoid bank conflicts.
#define P_LD   40                // halves per padded P row
#define K_LD   72                // halves per padded K row
#define V_LD   40                // halves per padded V row
#define P_OFF  0
#define K_OFF  (32 * P_LD)                  // 1280
#define K_BUF  (32 * K_LD)                  // 2304 halves per K buffer
#define V_OFF  (K_OFF + 2 * K_BUF)          // 5888
#define V_BUF  (64 * V_LD)                  // 2560 halves per V buffer
#define LDS_HALVES (V_OFF + 2 * V_BUF)      // 11008 halves (~21.5 KB)

__global__ __launch_bounds__(32) void attn_kernel(
    const _Float16* __restrict__ qh,   // [B,H,S,HD], pre-scaled by 1/sqrt(HD)
    const _Float16* __restrict__ kh,   // [B,H,S,HD]
    const _Float16* __restrict__ vT,   // [B,H,HD,S]
    _Float16* __restrict__ attn)       // [B,S,H,HD] == [MM, EE] row-major
{
  __shared__ _Float16 lds[LDS_HALVES];
  const int qi = blockIdx.x, h = blockIdx.y, b = blockIdx.z;
  const int lane = threadIdx.x & 31;
  const int r  = lane & 15;
  const int hf = lane >> 4;

  const size_t head = (size_t)(b * HH + h);
  const _Float16* qbase = qh + (head * SS + (size_t)qi * 32) * HD;
  const _Float16* kbase = kh + head * SS * HD;
  const _Float16* vbase = vT + head * HD * SS;
  const unsigned ldsbase = (unsigned)(size_t)(&lds[0]);   // LDS byte offset

  // Issue async copy of key-chunk jj (32 keys) into LDS buffer (jj & 1).
  auto issue_chunk = [&](int jj) {
    const int pc = jj & 1;
    const int nk = jj * 32;
    const char* gk  = (const char*)kbase + (size_t)(nk + lane) * (HD * 2);
    const char* gv0 = (const char*)vbase + ((size_t)lane * SS + nk) * 2;
    const char* gv1 = (const char*)vbase + ((size_t)(lane + 32) * SS + nk) * 2;
    unsigned dK  = ldsbase + (K_OFF + pc * K_BUF) * 2 + lane * (K_LD * 2);
    unsigned dV0 = ldsbase + (V_OFF + pc * V_BUF) * 2 + lane * (V_LD * 2);
    unsigned dV1 = dV0 + 32 * (V_LD * 2);
    // K chunk: 32 rows x 128B  (lane = key row)
    ASYNC_LD_B128(dK, gk, "0");   ASYNC_LD_B128(dK, gk, "16");
    ASYNC_LD_B128(dK, gk, "32");  ASYNC_LD_B128(dK, gk, "48");
    ASYNC_LD_B128(dK, gk, "64");  ASYNC_LD_B128(dK, gk, "80");
    ASYNC_LD_B128(dK, gk, "96");  ASYNC_LD_B128(dK, gk, "112");
    // V chunk: 64 rows x 64B    (lane = head-dim row, two row sets)
    ASYNC_LD_B128(dV0, gv0, "0"); ASYNC_LD_B128(dV0, gv0, "16");
    ASYNC_LD_B128(dV0, gv0, "32");ASYNC_LD_B128(dV0, gv0, "48");
    ASYNC_LD_B128(dV1, gv1, "0"); ASYNC_LD_B128(dV1, gv1, "16");
    ASYNC_LD_B128(dV1, gv1, "32");ASYNC_LD_B128(dV1, gv1, "48");
  };

  // Q fragments: 2 M-tiles x 2 K-steps (HD=64)
  v16h a_q[2][2];
#pragma unroll
  for (int mt = 0; mt < 2; ++mt)
#pragma unroll
    for (int kk = 0; kk < 2; ++kk)
      a_q[mt][kk] = load_frag(qbase + mt * 16 * HD + kk * 32, HD);

  v8f O[2][4];
  float mst[2][8], lst[2][8];
#pragma unroll
  for (int mt = 0; mt < 2; ++mt)
#pragma unroll
    for (int v = 0; v < 8; ++v) {
      mst[mt][v] = -1e30f;
      lst[mt][v] = 0.0f;
#pragma unroll
      for (int nd = 0; nd < 4; ++nd) O[mt][nd][v] = 0.0f;
    }

  issue_chunk(0);

  for (int j = 0; j <= qi; ++j) {
    const int p = j & 1;
    const _Float16* kbuf = &lds[K_OFF + p * K_BUF];
    const _Float16* vbuf = &lds[V_OFF + p * V_BUF];

    WAIT_ASYNC0();                 // chunk j resident in LDS
    if (j < qi) issue_chunk(j + 1);  // prefetch next chunk into other buffer

    // ---- scores S = Q @ K^T (32x32 tile), K frags from LDS ----
    v8f sacc[2][2];
#pragma unroll
    for (int mt = 0; mt < 2; ++mt)
#pragma unroll
      for (int nt = 0; nt < 2; ++nt)
#pragma unroll
        for (int v = 0; v < 8; ++v) sacc[mt][nt][v] = 0.0f;

#pragma unroll
    for (int nt = 0; nt < 2; ++nt) {
#pragma unroll
      for (int kk = 0; kk < 2; ++kk) {
        v16h bk = load_frag(kbuf + (size_t)(nt * 16) * K_LD + kk * 32, K_LD);
        sacc[0][nt] = wmma16(a_q[0][kk], bk, sacc[0][nt]);
        sacc[1][nt] = wmma16(a_q[1][kk], bk, sacc[1][nt]);
      }
    }

    // ---- online softmax; write P (f16) to LDS ----
#pragma unroll
    for (int mt = 0; mt < 2; ++mt) {
#pragma unroll
      for (int v = 0; v < 8; ++v) {
        const float s0 = sacc[mt][0][v];
        const float s1 = sacc[mt][1][v];
        const float rm = rmax16(fmaxf(s0, s1));
        const float mo = mst[mt][v];
        const float mn = fmaxf(mo, rm);
        const float cf = __expf(mo - mn);
        const float p0 = __expf(s0 - mn);
        const float p1 = __expf(s1 - mn);
        const float rs = rsum16(p0 + p1);
        lst[mt][v] = lst[mt][v] * cf + rs;
        mst[mt][v] = mn;
#pragma unroll
        for (int nd = 0; nd < 4; ++nd) O[mt][nd][v] *= cf;
        const int ml = mt * 16 + v + 8 * hf;
        lds[P_OFF + ml * P_LD + r]      = (_Float16)p0;
        lds[P_OFF + ml * P_LD + 16 + r] = (_Float16)p1;
      }
    }
    __syncthreads();

    // ---- O += P @ V  (K=32 keys, one WMMA K-step), frags from LDS ----
#pragma unroll
    for (int mt = 0; mt < 2; ++mt) {
      const int k0 = hf * 8;
      const _Float16* lp = &lds[P_OFF + (mt * 16 + r) * P_LD + k0];
      v8h lo = *(const v8h*)(lp);
      v8h hi = *(const v8h*)(lp + 16);
      v16h a_p = make_frag(lo, hi);
#pragma unroll
      for (int nd = 0; nd < 4; ++nd) {
        v16h bv = load_frag(vbuf + (size_t)(nd * 16) * V_LD, V_LD);
        O[mt][nd] = wmma16(a_p, bv, O[mt][nd]);
      }
    }
    __syncthreads();
  }

  // ---- normalize and store f16 [b, s, h, d] ----
#pragma unroll
  for (int mt = 0; mt < 2; ++mt) {
#pragma unroll
    for (int v = 0; v < 8; ++v) {
      const float inv = 1.0f / lst[mt][v];
      const int m = qi * 32 + mt * 16 + v + 8 * hf;
      const size_t row = ((size_t)b * SS + m) * EE + (size_t)h * HD;
#pragma unroll
      for (int nd = 0; nd < 4; ++nd)
        attn[row + nd * 16 + r] = (_Float16)(O[mt][nd][v] * inv);
    }
  }
}

// ---------- host ----------
extern "C" void kernel_launch(void* const* d_in, const int* in_sizes, int n_in,
                              void* d_out, int out_size, void* d_ws, size_t ws_size,
                              hipStream_t stream) {
  (void)in_sizes; (void)n_in; (void)out_size; (void)ws_size;
  const float* x  = (const float*)d_in[0];
  const float* Wq = (const float*)d_in[1];
  const float* bq = (const float*)d_in[2];
  const float* Wk = (const float*)d_in[3];
  const float* bk = (const float*)d_in[4];
  const float* Wv = (const float*)d_in[5];
  const float* bv = (const float*)d_in[6];
  const float* Wo = (const float*)d_in[7];
  const float* bo = (const float*)d_in[8];
  float* out = (float*)d_out;

  _Float16* x_h  = (_Float16*)d_ws;                       // [MM, EE]
  _Float16* wq_h = x_h  + (size_t)MM * EE;
  _Float16* wk_h = wq_h + (size_t)EE * EE;
  _Float16* wv_h = wk_h + (size_t)EE * EE;
  _Float16* wo_h = wv_h + (size_t)EE * EE;
  _Float16* q_h  = wo_h + (size_t)EE * EE;                // [B,H,S,HD]
  _Float16* k_h  = q_h  + (size_t)MM * EE;                // [B,H,S,HD]
  _Float16* vT_h = k_h  + (size_t)MM * EE;                // [B,H,HD,S]
  _Float16* at_h = vT_h + (size_t)MM * EE;                // [MM, EE]

  auto cvt = [&](const float* s, _Float16* d, int n) {
    int n4 = n / 4;
    cvt_f32_f16_kernel<<<dim3((n4 + 255) / 256), dim3(256), 0, stream>>>(s, d, n4);
  };
  cvt(x,  x_h,  MM * EE);
  cvt(Wq, wq_h, EE * EE);
  cvt(Wk, wk_h, EE * EE);
  cvt(Wv, wv_h, EE * EE);
  cvt(Wo, wo_h, EE * EE);

  const dim3 ggrid(MM / 32, EE / 64);
  const float qscale = 0.125f;  // 1/sqrt(HD)
  gemm_kernel<0><<<ggrid, 32, 0, stream>>>(x_h, wq_h, bq, q_h, nullptr, qscale);
  gemm_kernel<0><<<ggrid, 32, 0, stream>>>(x_h, wk_h, bk, k_h, nullptr, 1.0f);
  gemm_kernel<1><<<ggrid, 32, 0, stream>>>(x_h, wv_h, bv, vT_h, nullptr, 1.0f);

  attn_kernel<<<dim3(SS / 32, HH, BB), 32, 0, stream>>>(q_h, k_h, vT_h, at_h);

  gemm_kernel<2><<<ggrid, 32, 0, stream>>>(at_h, wo_h, bo, nullptr, out, 1.0f);
}
